// ControlPolicy_73632919322722
// MI455X (gfx1250) — matrix operand order repaired and checked by reading, not compiled, not run on hardware
//
#include <hip/hip_runtime.h>
#include <hip/hip_bf16.h>

typedef __attribute__((ext_vector_type(16))) _Float16 v16h;
typedef __attribute__((ext_vector_type(8)))  float    v8f;

#define TSTEPS 4096
#define BATCH  256
#define TWO_PI_F 6.28318530717958647692f

// ---------------------------------------------------------------------------
// Fast device math. Division via raw v_rcp_f32 (__builtin_amdgcn_rcpf) —
// avoids the IEEE div_scale/div_fmas expansion the compiler emits for '/'.
// ---------------------------------------------------------------------------
__device__ __forceinline__ float rcp_fast(float x) { return __builtin_amdgcn_rcpf(x); }
__device__ __forceinline__ float tanh_fast(float x) {
    // Pade 3/2 approx, clamped so it saturates at |x|>=3
    x = fminf(3.0f, fmaxf(-3.0f, x));
    float x2 = x * x;
    return x * (27.0f + x2) * rcp_fast(27.0f + 9.0f * x2);
}
__device__ __forceinline__ float sigmoid_(float x) { return rcp_fast(1.0f + __expf(-x)); }
__device__ __forceinline__ float softplus_(float x) { return log1pf(__expf(x)); }
__device__ __forceinline__ float bperm(int idx, float v) {
    return __int_as_float(__builtin_amdgcn_ds_bpermute(idx, __float_as_int(v)));
}

// ---------------------------------------------------------------------------
// Prep kernel: fold ln_w into a fused 64x48 weight matrix, emit it in the exact
// per-lane register order of the WMMA f16 B-matrix (32xK x 16xN tiles), and the
// ln_b correction vector c[j] = ln_b . W_col_j.
// Column map: j in [0,16)=w_state k0, [16,32)=w_state k1, 32/33=gate, 34=phase.
// B layout (16-bit B, 32x16): lane l=16h+n, element q <-> K=8h+(q&7)+16*(q>>3).
// ---------------------------------------------------------------------------
__global__ void prep_kernel(const float* __restrict__ ln_w,
                            const float* __restrict__ ln_b,
                            const float* __restrict__ w_state,
                            const float* __restrict__ gate_w,
                            const float* __restrict__ pos,
                            _Float16* __restrict__ Bh,
                            float* __restrict__ Cv) {
    int tid = threadIdx.x;
    auto wtab = [&](int nn, int j) -> float {
        if (j < 32) return w_state[(j >> 4) * 16 * 64 + (j & 15) * 64 + nn];
        if (j < 34) return gate_w[(j - 32) * 64 + nn];
        if (j == 34) return pos[nn];
        return 0.0f;
    };
    for (int idx = tid; idx < 3 * 2 * 32 * 16; idx += 256) {
        int q = idx & 15;
        int l = (idx >> 4) & 31;
        int s = (idx >> 9) & 1;
        int T = idx >> 10;
        int K = s * 32 + ((l >> 4) << 3) + (q & 7) + ((q >> 3) << 4);   // K index 0..63
        int j = T * 16 + (l & 15);              // output column 0..47
        Bh[idx] = (_Float16)(ln_w[K] * wtab(K, j));
    }
    if (tid < 48) {
        float acc = 0.0f;
        for (int nn = 0; nn < 64; ++nn) acc += ln_b[nn] * wtab(nn, tid);
        Cv[tid] = acc;
    }
}

// ---------------------------------------------------------------------------
// Main kernel: one wave32 per 16 batch rows, full T-scan in registers.
// Lane l = 16h + n.  z rows m=n split across lanes n (cols {0..7,16..23,32..39,
// 48..55}+8h pattern matching f16 A-matrix layout).  WMMA D tiles: lane holds
// D[row=v+8h][col=n] in VGPR v.
// ---------------------------------------------------------------------------
__global__ __launch_bounds__(32, 1) void policy_main(
    const float* __restrict__ states,
    const float* __restrict__ w_phase,
    const float* __restrict__ b_err,
    const float* __restrict__ kp_raw,
    const float* __restrict__ ki_raw,
    const float* __restrict__ kd_raw,
    const float* __restrict__ gate_b,
    const float* __restrict__ s_alpha,
    const float* __restrict__ s_leak,
    const float* __restrict__ s_beta,
    const float* __restrict__ s_rate,
    const float* __restrict__ s_aw,
    const float* __restrict__ s_omega,
    const _Float16* __restrict__ Bh,
    const float* __restrict__ Cv,
    float* __restrict__ out) {
    const int lane = threadIdx.x;
    const int n = lane & 15;
    const int h = lane >> 4;
    const int b0 = blockIdx.x * 16;

    // ---- scalar recurrence coefficients ----
    const float alpha = sigmoid_(s_alpha[0]);
    const float leak  = sigmoid_(s_leak[0]);
    const float beta  = sigmoid_(s_beta[0]);
    const float rate  = 0.25f * sigmoid_(s_rate[0]);
    const float aw    = softplus_(s_aw[0]);
    const float omega_base = softplus_(s_omega[0]) + 0.001f;
    const float oma = 1.0f - alpha, oml = 1.0f - leak, omb = 1.0f - beta;
    const float inv_rate = rcp_fast(rate + 1e-6f);

    // ---- per-lane (per action column n) constants ----
    const float kp0 = softplus_(kp_raw[n]),      kp1 = softplus_(kp_raw[16 + n]);
    const float ki0 = softplus_(ki_raw[n]),      ki1 = softplus_(ki_raw[16 + n]);
    const float kd0 = softplus_(kd_raw[n]),      kd1 = softplus_(kd_raw[16 + n]);
    const float be0 = b_err[n],                  be1 = b_err[16 + n];
    const float wps0 = w_phase[2 * n],           wpc0 = w_phase[2 * n + 1];
    const float wps1 = w_phase[2 * (16 + n)],    wpc1 = w_phase[2 * (16 + n) + 1];
    const float gb0 = gate_b[0], gb1 = gate_b[1];
    const float cs0 = Cv[n], cs1 = Cv[16 + n];
    const float cg0 = Cv[32], cg1 = Cv[33], cpos = Cv[34];

    // ---- WMMA B matrices (6 tiles of v16h) ----
    v16h Bt[3][2];
#pragma unroll
    for (int T = 0; T < 3; ++T)
#pragma unroll
        for (int s = 0; s < 2; ++s)
            Bt[T][s] = *(const v16h*)(Bh + ((T * 2 + s) * 32 + lane) * 16);

    // ---- carries ----
    float zr[32];
    float is0[8], is1[8], ep0[8], ep1[8], ap[8], phi[8];
#pragma unroll
    for (int i = 0; i < 32; ++i) zr[i] = 0.0f;
#pragma unroll
    for (int v = 0; v < 8; ++v) { is0[v] = is1[v] = ep0[v] = ep1[v] = ap[v] = phi[v] = 0.0f; }
    float s_t = 0.0f;

    const int pidx = (lane ^ 16) << 2;   // LN partner lane (byte index)
    const int bidx = (lane & 16) << 2;   // D2 column-0 lane of this half

    float* outb = out + (size_t)(b0 + 8 * h) * 16 + n;

    // per-lane x loader: 8x float4, matching f16 A-matrix K ordering
    auto loadx = [&](float (&xb)[32], int t) {
        const float* p = states + (size_t)t * (BATCH * 64) + (size_t)(b0 + n) * 64 + h * 8;
#pragma unroll
        for (int c = 0; c < 8; ++c) {
            float4 v4 = *(const float4*)(p + 16 * (c >> 1) + 4 * (c & 1));
            xb[4 * c + 0] = v4.x; xb[4 * c + 1] = v4.y;
            xb[4 * c + 2] = v4.z; xb[4 * c + 3] = v4.w;
        }
        // pull a future tile toward L2 (global_prefetch_b8)
        if (t + 16 < TSTEPS)
            __builtin_prefetch(p + (size_t)16 * (BATCH * 64), 0, 1);
    };

    auto step = [&](int t, const float (&x)[32]) {
        // ---- LayerNorm stats (row split across lanes n and n+16) ----
        float p1 = 0.0f, p2 = 0.0f;
#pragma unroll
        for (int i = 0; i < 32; ++i) { p1 += x[i]; p2 = fmaf(x[i], x[i], p2); }
        p1 += bperm(pidx, p1);
        p2 += bperm(pidx, p2);
        float mu = p1 * (1.0f / 64.0f);
        float var = p2 * (1.0f / 64.0f) - mu * mu;
        float am = alpha * rsqrtf(var + 1e-5f);
        // ---- EMA filter (ln_w/ln_b folded out) ----
#pragma unroll
        for (int i = 0; i < 32; ++i) zr[i] = oma * zr[i] + am * (x[i] - mu);
        s_t = oma * s_t + alpha;

        // ---- pack A (f16 16x32 layout), fused GEMM via WMMA ----
        v16h A0, A1;
#pragma unroll
        for (int i = 0; i < 16; ++i) {
            A0[i] = (_Float16)zr[i];
            A1[i] = (_Float16)zr[16 + i];
        }
        v8f Z = {};
        v8f D0 = __builtin_amdgcn_wmma_f32_16x16x32_f16(false, A0, false, Bt[0][0], (short)0, Z, false, false);
        D0     = __builtin_amdgcn_wmma_f32_16x16x32_f16(false, A1, false, Bt[0][1], (short)0, D0, false, false);
        v8f D1 = __builtin_amdgcn_wmma_f32_16x16x32_f16(false, A0, false, Bt[1][0], (short)0, Z, false, false);
        D1     = __builtin_amdgcn_wmma_f32_16x16x32_f16(false, A1, false, Bt[1][1], (short)0, D1, false, false);
        v8f D2 = __builtin_amdgcn_wmma_f32_16x16x32_f16(false, A0, false, Bt[2][0], (short)0, Z, false, false);
        D2     = __builtin_amdgcn_wmma_f32_16x16x32_f16(false, A1, false, Bt[2][1], (short)0, D2, false, false);

        // ---- per-row pointwise recurrences in D layout (row = v + 8h) ----
#pragma unroll
        for (int v = 0; v < 8; ++v) {
            // broadcast row scalars from D2 cols 0/1/2 (lanes 16h+0/1/2)
            float g0 = bperm(bidx,     D2[v]) + fmaf(s_t, cg0, gb0);
            float g1 = bperm(bidx + 4, D2[v]) + fmaf(s_t, cg1, gb1);
            float pd = bperm(bidx + 8, D2[v]) + s_t * cpos;
            float omega = omega_base + 0.02f * tanh_fast(pd);
            omega = fminf(1.0f, fmaxf(0.001f, omega));
            float ph = phi[v] + omega;
            ph = (ph >= TWO_PI_F) ? ph - TWO_PI_F : ph;
            phi[v] = ph;
            float sp = __sinf(ph), cp = __cosf(ph);
            float w0 = 0.5f + 0.5f * tanh_fast(0.5f * (g0 - g1));
            float w1 = 1.0f - w0;

            float e0 = D0[v] + fmaf(s_t, cs0, be0) + sp * wps0 + cp * wpc0;
            float e1 = D1[v] + fmaf(s_t, cs1, be1) + sp * wps1 + cp * wpc1;
            float i0n = oml * is0[v] + e0;
            float i1n = oml * is1[v] + e1;
            float u0 = kp0 * e0 + ki0 * i0n + kd0 * (e0 - ep0[v]);
            float u1 = kp1 * e1 + ki1 * i1n + kd1 * (e1 - ep1[v]);
            ep0[v] = e0; ep1[v] = e1;
            float u = w0 * u0 + w1 * u1;
            float us = tanh_fast(u);
            float alp = omb * ap[v] + beta * us;
            float d = rate * tanh_fast((alp - ap[v]) * inv_rate);
            float a = ap[v] + d;
            float awc = aw * (a - alp);
            is0[v] = i0n - awc;
            is1[v] = i1n - awc;
            ap[v] = a;
            outb[(size_t)t * (BATCH * 16) + v * 16] = a;
        }
    };

    // ---- software-pipelined scan: double-buffered x tiles ----
    float xa[32], xb[32];
    loadx(xa, 0);
    for (int t = 0; t < TSTEPS; t += 2) {
        loadx(xb, t + 1);
        step(t, xa);
        if (t + 2 < TSTEPS) loadx(xa, t + 2);
        step(t + 1, xb);
    }
}

// ---------------------------------------------------------------------------
extern "C" void kernel_launch(void* const* d_in, const int* in_sizes, int n_in,
                              void* d_out, int out_size, void* d_ws, size_t ws_size,
                              hipStream_t stream) {
    const float* states  = (const float*)d_in[0];
    const float* ln_w    = (const float*)d_in[1];
    const float* ln_b    = (const float*)d_in[2];
    const float* w_state = (const float*)d_in[3];
    const float* w_phase = (const float*)d_in[4];
    const float* b_err   = (const float*)d_in[5];
    const float* kp_raw  = (const float*)d_in[6];
    const float* ki_raw  = (const float*)d_in[7];
    const float* kd_raw  = (const float*)d_in[8];
    const float* gate_w  = (const float*)d_in[9];
    const float* gate_b  = (const float*)d_in[10];
    const float* f_alpha = (const float*)d_in[11];
    const float* f_leak  = (const float*)d_in[12];
    const float* f_beta  = (const float*)d_in[13];
    const float* f_rate  = (const float*)d_in[14];
    const float* f_aw    = (const float*)d_in[15];
    const float* f_omega = (const float*)d_in[16];
    const float* pos     = (const float*)d_in[17];

    _Float16* Bh = (_Float16*)d_ws;
    float* Cv = (float*)((char*)d_ws + 3 * 2 * 32 * 16 * sizeof(_Float16));

    hipLaunchKernelGGL(prep_kernel, dim3(1), dim3(256), 0, stream,
                       ln_w, ln_b, w_state, gate_w, pos, Bh, Cv);
    hipLaunchKernelGGL(policy_main, dim3(16), dim3(32), 0, stream,
                       states, w_phase, b_err, kp_raw, ki_raw, kd_raw, gate_b,
                       f_alpha, f_leak, f_beta, f_rate, f_aw, f_omega,
                       (const _Float16*)Bh, (const float*)Cv, (float*)d_out);
}
